// Gemma4SlidingAttention_47115791237726
// MI455X (gfx1250) — matrix operand order repaired and checked by reading, not compiled, not run on hardware
//
#include <hip/hip_runtime.h>
#include <hip/hip_bf16.h>

// ---------------------------------------------------------------------------
// Gemma sliding attention for MI455X (gfx1250), bf16 WMMA pipeline.
// ---------------------------------------------------------------------------

typedef __attribute__((ext_vector_type(16))) __bf16 v16bf;
typedef __attribute__((ext_vector_type(8)))  __bf16 v8bf;
typedef __attribute__((ext_vector_type(4)))  __bf16 v4bf;
typedef __attribute__((ext_vector_type(8)))  float  v8f;

#define HIDDEN 2048
#define NH     16
#define NKV    8
#define HD     128
#define WINDOW 1024
#define BB     2
#define LL     2048
#define EPSV   1e-6f
#define SCALE  0.08838834764831845f   // 128^-0.5

#if __has_builtin(__builtin_amdgcn_global_load_async_to_lds_b128) && \
    __has_builtin(__builtin_amdgcn_s_wait_asynccnt)
#define HAVE_ASYNC_LDS 1
// exact parameter types per hipcc diagnostic: (v4i __device__*, v4i __shared__*, Ii, Ii)
typedef int v4i __attribute__((vector_size(16)));
typedef __attribute__((address_space(1))) v4i* gv4i_p;   // global
typedef __attribute__((address_space(3))) v4i* lv4i_p;   // LDS
#else
#define HAVE_ASYNC_LDS 0
#endif

// f32 -> bf16 via native conversion (RTNE on CDNA5)
__device__ __forceinline__ __bf16 f2bf(float f) { return (__bf16)f; }

__device__ __forceinline__ v8bf cvt8(const float4& a, const float4& b) {
  v8bf r;
  r[0] = (__bf16)a.x; r[1] = (__bf16)a.y; r[2] = (__bf16)a.z; r[3] = (__bf16)a.w;
  r[4] = (__bf16)b.x; r[5] = (__bf16)b.y; r[6] = (__bf16)b.z; r[7] = (__bf16)b.w;
  return r;
}

__device__ __forceinline__ v16bf make_frag(const __bf16* lo, const __bf16* hi) {
  v8bf a = *reinterpret_cast<const v8bf*>(lo);
  v8bf b = *reinterpret_cast<const v8bf*>(hi);
  v16bf r;
#pragma unroll
  for (int i = 0; i < 8; ++i) { r[i] = a[i]; r[i + 8] = b[i]; }
  return r;
}

__device__ __forceinline__ v8f vzero8() {
  v8f z;
#pragma unroll
  for (int i = 0; i < 8; ++i) z[i] = 0.0f;
  return z;
}

// ---------------------------------------------------------------------------
// Generic GEMM: C[M,N] = A[M,K] @ W[N,K]^T  (W row-major [N,K], f32)
// A is f32 (projections) or bf16 (attention output) selected by template.
// Block: 256 threads = 8 waves; block tile 128x128; wave tile 32x64; K step 32.
// ---------------------------------------------------------------------------
template <bool ABF>
__global__ __launch_bounds__(256) void gemm_bf16_wmma(
    const void* __restrict__ Ap, const float* __restrict__ W,
    float* __restrict__ C, int M, int N, int K) {
  __shared__ __bf16 sA[128][40];
  __shared__ __bf16 sW[128][40];

  const int tid  = threadIdx.x;
  const int lane = tid & 31;
  const int hf   = lane >> 4;      // 16-lane half
  const int ln   = lane & 15;
  const int wv   = tid >> 5;       // wave 0..7
  const int wr   = wv & 3;         // wave row 0..3  (32 rows each)
  const int wc   = wv >> 2;        // wave col 0..1  (64 cols each)
  const int bm   = blockIdx.y * 128;
  const int bn   = blockIdx.x * 128;

  v8f acc[2][4];
#pragma unroll
  for (int i = 0; i < 2; ++i)
#pragma unroll
    for (int j = 0; j < 4; ++j) acc[i][j] = vzero8();

  const int arow = tid >> 1;           // 0..127
  const int acol = (tid & 1) * 16;     // 0 or 16

  for (int k0 = 0; k0 < K; k0 += 32) {
    // ---- stage A tile (128x32) ----
    if (ABF) {
      const __bf16* A = (const __bf16*)Ap;
      const __bf16* src = A + (size_t)(bm + arow) * K + k0 + acol;
      *reinterpret_cast<v8bf*>(&sA[arow][acol])     = *reinterpret_cast<const v8bf*>(src);
      *reinterpret_cast<v8bf*>(&sA[arow][acol + 8]) = *reinterpret_cast<const v8bf*>(src + 8);
      if (k0 + 32 < K) __builtin_prefetch(src + 32, 0, 0);
    } else {
      const float* A = (const float*)Ap;
      const float* src = A + (size_t)(bm + arow) * K + k0 + acol;
      float4 a0 = *reinterpret_cast<const float4*>(src);
      float4 a1 = *reinterpret_cast<const float4*>(src + 4);
      float4 a2 = *reinterpret_cast<const float4*>(src + 8);
      float4 a3 = *reinterpret_cast<const float4*>(src + 12);
      *reinterpret_cast<v8bf*>(&sA[arow][acol])     = cvt8(a0, a1);
      *reinterpret_cast<v8bf*>(&sA[arow][acol + 8]) = cvt8(a2, a3);
      if (k0 + 32 < K) __builtin_prefetch(src + 32, 0, 0);
    }
    // ---- stage W tile (128x32), f32 -> bf16 ----
    {
      const float* src = W + (size_t)(bn + arow) * K + k0 + acol;
      float4 w0 = *reinterpret_cast<const float4*>(src);
      float4 w1 = *reinterpret_cast<const float4*>(src + 4);
      float4 w2 = *reinterpret_cast<const float4*>(src + 8);
      float4 w3 = *reinterpret_cast<const float4*>(src + 12);
      *reinterpret_cast<v8bf*>(&sW[arow][acol])     = cvt8(w0, w1);
      *reinterpret_cast<v8bf*>(&sW[arow][acol + 8]) = cvt8(w2, w3);
      if (k0 + 32 < K) __builtin_prefetch(src + 32, 0, 0);
    }
    __syncthreads();

    v16bf af[2], bfr[4];
#pragma unroll
    for (int mg = 0; mg < 2; ++mg) {
      const __bf16* p = &sA[wr * 32 + mg * 16 + ln][0];
      af[mg] = make_frag(p + hf * 8, p + 16 + hf * 8);
    }
#pragma unroll
    for (int nf = 0; nf < 4; ++nf) {
      const __bf16* p = &sW[wc * 64 + nf * 16 + ln][0];
      bfr[nf] = make_frag(p + hf * 8, p + 16 + hf * 8);
    }
#pragma unroll
    for (int mg = 0; mg < 2; ++mg)
#pragma unroll
      for (int nf = 0; nf < 4; ++nf)
        acc[mg][nf] = __builtin_amdgcn_wmma_f32_16x16x32_bf16(
            false, af[mg], false, bfr[nf], (short)0, acc[mg][nf], false, false);
    __syncthreads();
  }

#pragma unroll
  for (int mg = 0; mg < 2; ++mg)
#pragma unroll
    for (int nf = 0; nf < 4; ++nf)
#pragma unroll
      for (int r = 0; r < 8; ++r) {
        int row = bm + wr * 32 + mg * 16 + r + hf * 8;
        int col = bn + wc * 64 + nf * 16 + ln;
        C[(size_t)row * N + col] = acc[mg][nf][r];
      }
}

// ---------------------------------------------------------------------------
// Per-(b,l,head) RMS norm (optional) + relayout to head-major bf16.
// One wave (32 threads) per 128-element row.
// X: [b][l][head][128] f32 ; Y: [(b*heads+head)*L + l][128] bf16
// ---------------------------------------------------------------------------
__global__ __launch_bounds__(32) void norm_relayout(
    const float* __restrict__ X, const float* __restrict__ w,
    __bf16* __restrict__ Y, int heads, int do_norm) {
  const int bid  = blockIdx.x;
  const int head = bid % heads;
  const int l    = (bid / heads) % LL;
  const int b    = bid / (heads * LL);
  const int lane = threadIdx.x;

  const float4 x = *reinterpret_cast<const float4*>(X + (size_t)bid * HD + lane * 4);
  float ss = x.x * x.x + x.y * x.y + x.z * x.z + x.w * x.w;
#pragma unroll
  for (int off = 16; off > 0; off >>= 1) ss += __shfl_xor(ss, off, 32);

  float sc = do_norm ? rsqrtf(ss * (1.0f / HD) + EPSV) : 1.0f;
  const float4 wv4 = *reinterpret_cast<const float4*>(w + lane * 4);
  v4bf o;
  o[0] = f2bf(x.x * sc * (do_norm ? wv4.x : 1.0f));
  o[1] = f2bf(x.y * sc * (do_norm ? wv4.y : 1.0f));
  o[2] = f2bf(x.z * sc * (do_norm ? wv4.z : 1.0f));
  o[3] = f2bf(x.w * sc * (do_norm ? wv4.w : 1.0f));
  size_t out = ((size_t)(b * heads + head) * LL + l) * HD + lane * 4;
  *reinterpret_cast<v4bf*>(Y + out) = o;
}

// ---------------------------------------------------------------------------
// Flash-style masked attention.
// Grid: (L/128, B*NH). Block: 256 threads = 8 waves; each wave owns 16 q-rows.
// Mask (faithful to reference, L>WINDOW): -inf iff col>=L-WINDOW && col<=row.
// Output: ob[(b*L+l)*NH*HD + h*HD + d] bf16 (ready for O projection).
// ---------------------------------------------------------------------------
__global__ __launch_bounds__(256) void attn_sw(
    const __bf16* __restrict__ qb, const __bf16* __restrict__ kb,
    const __bf16* __restrict__ vb, __bf16* __restrict__ ob) {
  __shared__ __bf16 sK[64][136];       // key tile  [key][d]
  __shared__ __bf16 sVT[128][72];      // V tile transposed [d][key]
  __shared__ __bf16 sP[8][16][72];     // per-wave probs [row][key]

  const int tid  = threadIdx.x;
  const int lane = tid & 31;
  const int hf   = lane >> 4;
  const int ln   = lane & 15;
  const int wv   = tid >> 5;
  const int bh   = blockIdx.y;
  const int b    = bh >> 4;
  const int h    = bh & 15;
  const int kv   = h >> 1;                         // GQA n_rep = 2
  const int q0   = blockIdx.x * 128 + wv * 16;     // first q row of this wave

  // Q fragments for the wave's 16 rows (resident for the whole kernel)
  v16bf qf[4];
  {
    const __bf16* qptr = qb + ((size_t)(b * NH + h) * LL + q0 + ln) * HD;
#pragma unroll
    for (int kq = 0; kq < 4; ++kq)
      qf[kq] = make_frag(qptr + kq * 32 + hf * 8, qptr + kq * 32 + 16 + hf * 8);
  }

  v8f oacc[8];
#pragma unroll
  for (int i = 0; i < 8; ++i) oacc[i] = vzero8();
  float m[8], ls[8];
#pragma unroll
  for (int r = 0; r < 8; ++r) { m[r] = -INFINITY; ls[r] = 0.0f; }

  const __bf16* kbase = kb + (size_t)(b * NKV + kv) * LL * HD;
  const __bf16* vbase = vb + (size_t)(b * NKV + kv) * LL * HD;

  // staging maps
  const int krow = tid >> 2;            // 0..63, K tile row
  const int kch  = (tid & 3) * 32;      // 0/32/64/96
  const int vq4  = (tid >> 4) * 4;      // V: 4 key rows starting here (0..60)
  const int vdch = (tid & 15) * 8;      // V: 8 d-values starting here (0..120)

  for (int kt = 0; kt < LL / 64; ++kt) {
    const int key0 = kt * 64;

    // ---- stage K tile (global -> LDS, async when available) ----
    {
      const __bf16* ksrc = kbase + (size_t)(key0 + krow) * HD + kch;
#if HAVE_ASYNC_LDS
#pragma unroll
      for (int i = 0; i < 4; ++i)
        __builtin_amdgcn_global_load_async_to_lds_b128(
            (gv4i_p)(uintptr_t)(ksrc + i * 8),
            (lv4i_p)(uintptr_t)(&sK[krow][kch + i * 8]),
            0, 0);
#else
#pragma unroll
      for (int i = 0; i < 4; ++i)
        *reinterpret_cast<v8bf*>(&sK[krow][kch + i * 8]) =
            *reinterpret_cast<const v8bf*>(ksrc + i * 8);
#endif
      if (kt + 1 < LL / 64) __builtin_prefetch(ksrc + 64 * HD, 0, 0);
    }
    // ---- stage V tile transposed: 4 key rows x 8 d per thread ----
    {
      v8bf vr[4];
#pragma unroll
      for (int i = 0; i < 4; ++i)
        vr[i] = *reinterpret_cast<const v8bf*>(
            vbase + (size_t)(key0 + vq4 + i) * HD + vdch);
#pragma unroll
      for (int e = 0; e < 8; ++e) {
        v4bf pk;
#pragma unroll
        for (int i = 0; i < 4; ++i) pk[i] = vr[i][e];
        *reinterpret_cast<v4bf*>(&sVT[vdch + e][vq4]) = pk;
      }
      if (kt + 1 < LL / 64)
        __builtin_prefetch(vbase + (size_t)(key0 + 64 + vq4) * HD + vdch, 0, 0);
    }
#if HAVE_ASYNC_LDS
    __builtin_amdgcn_s_wait_asynccnt(0);
#endif
    __syncthreads();

    // ---- S = Q @ K^T (16x64 per wave) ----
    v8f s[4];
#pragma unroll
    for (int i = 0; i < 4; ++i) s[i] = vzero8();
#pragma unroll
    for (int kq = 0; kq < 4; ++kq) {
#pragma unroll
      for (int nf = 0; nf < 4; ++nf) {
        const __bf16* p = &sK[nf * 16 + ln][kq * 32];
        v16bf bfr = make_frag(p + hf * 8, p + 16 + hf * 8);
        s[nf] = __builtin_amdgcn_wmma_f32_16x16x32_bf16(
            false, qf[kq], false, bfr, (short)0, s[nf], false, false);
      }
    }

    // ---- scale + mask + row max ----
    float rmax[8];
#pragma unroll
    for (int r = 0; r < 8; ++r) rmax[r] = -INFINITY;
#pragma unroll
    for (int nf = 0; nf < 4; ++nf) {
      const int col = key0 + nf * 16 + ln;
#pragma unroll
      for (int r = 0; r < 8; ++r) {
        const int row = q0 + r + hf * 8;
        float sv = s[nf][r] * SCALE;
        if (col >= (LL - WINDOW) && col <= row) sv = -INFINITY;
        s[nf][r] = sv;
        rmax[r] = fmaxf(rmax[r], sv);
      }
    }
#pragma unroll
    for (int r = 0; r < 8; ++r)
#pragma unroll
      for (int off = 8; off >= 1; off >>= 1)
        rmax[r] = fmaxf(rmax[r], __shfl_xor(rmax[r], off, 32));

    // ---- online softmax ----
    float fac[8], rsum[8];
#pragma unroll
    for (int r = 0; r < 8; ++r) {
      const float mn = fmaxf(m[r], rmax[r]);
      fac[r]  = (mn == -INFINITY) ? 0.0f : __expf(m[r] - mn);
      m[r]    = mn;
      rsum[r] = 0.0f;
    }
#pragma unroll
    for (int nf = 0; nf < 4; ++nf)
#pragma unroll
      for (int r = 0; r < 8; ++r) {
        const float p = (m[r] == -INFINITY) ? 0.0f : __expf(s[nf][r] - m[r]);
        rsum[r] += p;
        sP[wv][r + hf * 8][nf * 16 + ln] = f2bf(p);
      }
#pragma unroll
    for (int r = 0; r < 8; ++r) {
#pragma unroll
      for (int off = 8; off >= 1; off >>= 1) rsum[r] += __shfl_xor(rsum[r], off, 32);
      ls[r] = ls[r] * fac[r] + rsum[r];
    }
#pragma unroll
    for (int nb = 0; nb < 8; ++nb)
#pragma unroll
      for (int r = 0; r < 8; ++r) oacc[nb][r] *= fac[r];

    // ---- O += P @ V ----
#pragma unroll
    for (int kp = 0; kp < 2; ++kp) {
      const __bf16* pp = &sP[wv][ln][kp * 32];
      v16bf pf = make_frag(pp + hf * 8, pp + 16 + hf * 8);
#pragma unroll
      for (int nb = 0; nb < 8; ++nb) {
        const __bf16* vp = &sVT[nb * 16 + ln][kp * 32];
        v16bf vf = make_frag(vp + hf * 8, vp + 16 + hf * 8);
        oacc[nb] = __builtin_amdgcn_wmma_f32_16x16x32_bf16(
            false, pf, false, vf, (short)0, oacc[nb], false, false);
      }
    }
    __syncthreads();
  }

  // ---- epilogue: normalize and store head-interleaved bf16 ----
#pragma unroll
  for (int nb = 0; nb < 8; ++nb)
#pragma unroll
    for (int r = 0; r < 8; ++r) {
      const int lrow = q0 + r + hf * 8;
      const int d    = nb * 16 + ln;
      const float inv = 1.0f / ls[r];
      ob[((size_t)(b * LL + lrow)) * (NH * HD) + h * HD + d] = f2bf(oacc[nb][r] * inv);
    }
}

// ---------------------------------------------------------------------------
extern "C" void kernel_launch(void* const* d_in, const int* in_sizes, int n_in,
                              void* d_out, int out_size, void* d_ws, size_t ws_size,
                              hipStream_t stream) {
  const float* hidden = (const float*)d_in[0];  // [B,L,2048]
  const float* Wq     = (const float*)d_in[1];  // [2048,2048]
  const float* Wk     = (const float*)d_in[2];  // [1024,2048]
  const float* Wv     = (const float*)d_in[3];  // [1024,2048]
  const float* Wo     = (const float*)d_in[4];  // [2048,2048]
  const float* qnw    = (const float*)d_in[5];  // [128]
  const float* knw    = (const float*)d_in[6];  // [128]
  float* out = (float*)d_out;

  const int M = BB * LL;  // 4096
  // workspace partition (256B aligned)
  size_t off = 0;
  auto alloc = [&](size_t bytes) {
    void* p = (char*)d_ws + off;
    off += (bytes + 255) & ~(size_t)255;
    return p;
  };
  float*  qf32 = (float*) alloc((size_t)M * 2048 * 4);
  float*  kf32 = (float*) alloc((size_t)M * 1024 * 4);
  float*  vf32 = (float*) alloc((size_t)M * 1024 * 4);
  __bf16* qb   = (__bf16*)alloc((size_t)M * 2048 * 2);
  __bf16* kb   = (__bf16*)alloc((size_t)M * 1024 * 2);
  __bf16* vb   = (__bf16*)alloc((size_t)M * 1024 * 2);
  __bf16* ob   = (__bf16*)alloc((size_t)M * 2048 * 2);
  (void)ws_size; (void)n_in; (void)in_sizes; (void)out_size;

  // 1) projections
  gemm_bf16_wmma<false><<<dim3(2048 / 128, M / 128), 256, 0, stream>>>(
      hidden, Wq, qf32, M, 2048, HIDDEN);
  gemm_bf16_wmma<false><<<dim3(1024 / 128, M / 128), 256, 0, stream>>>(
      hidden, Wk, kf32, M, 1024, HIDDEN);
  gemm_bf16_wmma<false><<<dim3(1024 / 128, M / 128), 256, 0, stream>>>(
      hidden, Wv, vf32, M, 1024, HIDDEN);

  // 2) RMS norm (+relayout to head-major bf16); V is convert-only
  norm_relayout<<<BB * LL * NH, 32, 0, stream>>>(qf32, qnw, qb, NH, 1);
  norm_relayout<<<BB * LL * NKV, 32, 0, stream>>>(kf32, knw, kb, NKV, 1);
  norm_relayout<<<BB * LL * NKV, 32, 0, stream>>>(vf32, qnw, vb, NKV, 0);

  // 3) masked flash attention
  attn_sw<<<dim3(LL / 128, BB * NH), 256, 0, stream>>>(qb, kb, vb, ob);

  // 4) output projection
  gemm_bf16_wmma<true><<<dim3(2048 / 128, M / 128), 256, 0, stream>>>(
      ob, Wo, out, M, 2048, HIDDEN);
}
